// GNNLayer_26139170964197
// MI455X (gfx1250) — matrix operand order repaired and checked by reading, not compiled, not run on hardware
//
#include <hip/hip_runtime.h>
#include <hip/hip_bf16.h>

#define B_ 16
#define N_ 10000
#define D_ 128
#define TILE_ 64
#define NTILES_PB ((N_ + TILE_ - 1) / TILE_)   // 157

typedef _Float16 v8h  __attribute__((ext_vector_type(8)));
typedef _Float16 v16h __attribute__((ext_vector_type(16)));
typedef float    v8f  __attribute__((ext_vector_type(8)));

// ---------------------------------------------------------------------------
// Weight pre-pack: f32 row-major (K x 128) -> f16 fragment-major layout that
// matches the CDNA5 WMMA 32x16 f16 B-operand lane layout:
//   frag[(kstep*8 + nt)*32 + lane][j] = W[kstep*32 + (lane>=16?8:0) + (j&7) + (j>=8?16:0)][nt*16 + (lane&15)]
// so the main kernel's B load is a single aligned 32B (v16h) load per lane.
// ---------------------------------------------------------------------------
__global__ void pack_w(const float* __restrict__ W, _Float16* __restrict__ dst, int total) {
    int tid = blockIdx.x * blockDim.x + threadIdx.x;
    if (tid >= total) return;
    int j     = tid & 15;
    int lane  = (tid >> 4) & 31;
    int nt    = (tid >> 9) & 7;
    int kstep = tid >> 12;
    int k = kstep * 32 + ((lane >> 4) << 3) + (j & 7) + ((j >> 3) << 4);
    int n = (nt << 4) + (lane & 15);
    dst[tid] = (_Float16)W[k * D_ + n];
}

// ---------------------------------------------------------------------------
// WMMA helpers
// ---------------------------------------------------------------------------
__device__ inline v8f wmma_f16(v16h a, v16h b, v8f c) {
    // D = A(16x32 f16) * B(32x16 f16) + C(16x16 f32)
    return __builtin_amdgcn_wmma_f32_16x16x32_f16(false, a, false, b, (short)0, c, false, false);
}

// A-operand load from row-major f16 LDS buffer (row stride = 128 halves).
// ISA 16-bit A 16x32 layout: lanes 0-15 rows M=0..15 (K groups 0-7,16-23),
// lanes 16-31 same rows (K groups 8-15,24-31).
__device__ inline v16h load_a(const _Float16* X, int rowBase, int k0, int lane) {
    int row = rowBase + (lane & 15);
    int kg  = (lane >> 4) << 3;
    const _Float16* p = X + row * D_ + k0 + kg;
    v8h lo = *(const v8h*)(p);
    v8h hi = *(const v8h*)(p + 16);
    return __builtin_shufflevector(lo, hi, 0,1,2,3,4,5,6,7,8,9,10,11,12,13,14,15);
}

// B-operand load from fragment-major packed weights (one 32B load per lane).
__device__ inline v16h load_b(const _Float16* __restrict__ Wf, int kstep, int nt, int lane) {
    return *(const v16h*)(Wf + (size_t)((((kstep << 3) + nt) << 5) + lane) * 16);
}

// SiLU with native v_rcp_f32 (result feeds an f16 operand -> ~1 ULP rcp is plenty;
// avoids the ~10-instruction IEEE divide expansion seen in the previous build).
__device__ inline float silu_f(float x) {
    return x * __builtin_amdgcn_rcpf(1.f + __expf(-x));
}

// ---------------------------------------------------------------------------
// Main fused kernel: gather -> 2x message MLP -> update MLP -> LN -> scatter
// ---------------------------------------------------------------------------
__global__ __launch_bounds__(256, 2) void gnn_layer(
    const float* __restrict__ h, const int* __restrict__ tour,
    const _Float16* __restrict__ W1m_f, const float* __restrict__ b1m,
    const _Float16* __restrict__ W2m_f, const float* __restrict__ b2m,
    const _Float16* __restrict__ W1u_f, const float* __restrict__ b1u,
    const _Float16* __restrict__ W2u_f, const float* __restrict__ b2u,
    const float* __restrict__ gamma, const float* __restrict__ beta,
    float* __restrict__ out)
{
    extern __shared__ char smem[];
    _Float16* xh   = (_Float16*)(smem);           // 66 x 128 f16 halo rows (16896 B)
    _Float16* yA   = (_Float16*)(smem + 16896);   // 64 x 128 f16 scratch   (16384 B)
    _Float16* yB   = (_Float16*)(smem + 33280);   // 64 x 128 f16 scratch   (16384 B)
    float*    posb = (float*)   (smem + 49664);   // 64 x 128 f32 residual  (32768 B)
    float*    red  = (float*)   (smem + 82432);   // 64 x 8  f32 LN reduce  ( 2048 B)

    const int tid  = threadIdx.x;
    const int lane = tid & 31;
    const int wave = tid >> 5;
    const int b    = blockIdx.x / NTILES_PB;
    const int i0   = (blockIdx.x % NTILES_PB) * TILE_;
    const int validRows = (N_ - i0 < TILE_) ? (N_ - i0) : TILE_;

    // ---- gather: halo rows hr=0..65 correspond to tour positions i0-1..i0+64 (mod N)
    {
        const int q = tid & 3;  // quarter of a 128-float row
        for (int hr = tid >> 2; hr < TILE_ + 2; hr += 64) {
            int p = (i0 + hr - 1 + N_) % N_;
            int node = tour[b * N_ + p];
            const float4* s4 = (const float4*)(h + ((size_t)b * N_ + (size_t)node) * D_ + q * 32);
            _Float16* dh = xh + hr * D_ + q * 32;
            float4* df = (hr >= 1 && hr <= TILE_) ? (float4*)(posb + (hr - 1) * D_ + q * 32)
                                                  : (float4*)nullptr;
            #pragma unroll
            for (int k = 0; k < 8; ++k) {
                float4 v = s4[k];
                dh[4*k+0] = (_Float16)v.x; dh[4*k+1] = (_Float16)v.y;
                dh[4*k+2] = (_Float16)v.z; dh[4*k+3] = (_Float16)v.w;
                if (df) df[k] = v;
            }
        }
    }
    __syncthreads();

    const int mBase  = (wave >> 1) << 4;   // which 16-row m-tile this wave owns
    const int ntBase = (wave & 1) << 2;    // which 4 n-tiles (of 8) this wave owns
    const int colLo  = lane & 15;
    const int rOff   = (lane >> 4) << 3;   // C/D layout: lanes 16-31 hold rows M=8..15

    // ================= message MLP: msg = mlp([pos|prev]) + mlp([pos|next]) =====
    v8f msg[4] = {};
    #pragma unroll 1
    for (int pass = 0; pass < 2; ++pass) {
        const int hiBase = (pass == 0) ? 0 : 2;  // prev rows / next rows in halo buffer
        v8f acc[4] = {};
        #pragma unroll 1
        for (int ks = 0; ks < 8; ++ks) {         // K = 256 (cat of pos | neighbor)
            v16h a = (ks < 4) ? load_a(xh, 1 + mBase, ks * 32, lane)
                              : load_a(xh, hiBase + mBase, (ks - 4) * 32, lane);
            #pragma unroll
            for (int t = 0; t < 4; ++t)
                acc[t] = wmma_f16(a, load_b(W1m_f, ks, ntBase + t, lane), acc[t]);
        }
        #pragma unroll
        for (int t = 0; t < 4; ++t) {           // + b1m, SiLU -> yA (f16)
            int col = ((ntBase + t) << 4) + colLo;
            float bv = b1m[col];
            #pragma unroll
            for (int r = 0; r < 8; ++r)
                yA[(mBase + rOff + r) * D_ + col] = (_Float16)silu_f(acc[t][r] + bv);
        }
        __syncthreads();
        #pragma unroll 1
        for (int ks = 0; ks < 4; ++ks) {        // second linear, accumulate into msg
            v16h a = load_a(yA, mBase, ks * 32, lane);
            #pragma unroll
            for (int t = 0; t < 4; ++t)
                msg[t] = wmma_f16(a, load_b(W2m_f, ks, ntBase + t, lane), msg[t]);
        }
        __syncthreads();                        // yA is rewritten next pass
    }
    #pragma unroll
    for (int t = 0; t < 4; ++t) {               // msg + 2*b2m -> yB (f16)
        int col = ((ntBase + t) << 4) + colLo;
        float bv = 2.f * b2m[col];
        #pragma unroll
        for (int r = 0; r < 8; ++r)
            yB[(mBase + rOff + r) * D_ + col] = (_Float16)(msg[t][r] + bv);
    }
    __syncthreads();

    // ================= update MLP: u = mlp_u([pos|msg]) ==========================
    {
        v8f acc[4] = {};
        #pragma unroll 1
        for (int ks = 0; ks < 8; ++ks) {        // K = 256 (cat of pos | msg)
            v16h a = (ks < 4) ? load_a(xh, 1 + mBase, ks * 32, lane)
                              : load_a(yB, mBase, (ks - 4) * 32, lane);
            #pragma unroll
            for (int t = 0; t < 4; ++t)
                acc[t] = wmma_f16(a, load_b(W1u_f, ks, ntBase + t, lane), acc[t]);
        }
        #pragma unroll
        for (int t = 0; t < 4; ++t) {           // + b1u, SiLU -> yA
            int col = ((ntBase + t) << 4) + colLo;
            float bv = b1u[col];
            #pragma unroll
            for (int r = 0; r < 8; ++r)
                yA[(mBase + rOff + r) * D_ + col] = (_Float16)silu_f(acc[t][r] + bv);
        }
        __syncthreads();
        v8f acc2[4] = {};
        #pragma unroll 1
        for (int ks = 0; ks < 4; ++ks) {
            v16h a = load_a(yA, mBase, ks * 32, lane);
            #pragma unroll
            for (int t = 0; t < 4; ++t)
                acc2[t] = wmma_f16(a, load_b(W2u_f, ks, ntBase + t, lane), acc2[t]);
        }
        #pragma unroll
        for (int t = 0; t < 4; ++t) {           // residual: posb = h_pos + u
            int col = ((ntBase + t) << 4) + colLo;
            float bv = b2u[col];
            #pragma unroll
            for (int r = 0; r < 8; ++r) {
                int row = mBase + rOff + r;
                posb[row * D_ + col] += acc2[t][r] + bv;
            }
        }
    }
    __syncthreads();

    // ================= LayerNorm + scatter ======================================
    {
        const int r = tid >> 2, q = tid & 3;
        const float* rp = posb + r * D_ + q * 32;
        float s = 0.f, ss = 0.f;
        #pragma unroll
        for (int c = 0; c < 32; ++c) { float v = rp[c]; s += v; ss += v * v; }
        red[r * 8 + q]     = s;
        red[r * 8 + 4 + q] = ss;
        __syncthreads();
        if (q == 0) {
            float sum = red[r*8+0] + red[r*8+1] + red[r*8+2] + red[r*8+3];
            float sq  = red[r*8+4] + red[r*8+5] + red[r*8+6] + red[r*8+7];
            float mu  = sum * (1.f / 128.f);
            float var = sq * (1.f / 128.f) - mu * mu;
            red[r*8+0] = mu;
            red[r*8+1] = rsqrtf(var + 1e-5f);
        }
        __syncthreads();
        if (r < validRows) {
            float mu = red[r*8+0], ri = red[r*8+1];
            int node = tour[b * N_ + i0 + r];
            float* op = out + ((size_t)b * N_ + (size_t)node) * D_ + q * 32;
            #pragma unroll
            for (int c = 0; c < 32; ++c) {
                int col = q * 32 + c;
                op[c] = (rp[c] - mu) * ri * gamma[col] + beta[col];
            }
        }
    }
}

// ---------------------------------------------------------------------------
extern "C" void kernel_launch(void* const* d_in, const int* in_sizes, int n_in,
                              void* d_out, int out_size, void* d_ws, size_t ws_size,
                              hipStream_t stream) {
    const float* h     = (const float*)d_in[0];
    const int*   tour  = (const int*)  d_in[1];
    const float* W1m   = (const float*)d_in[2];
    const float* b1m   = (const float*)d_in[3];
    const float* W2m   = (const float*)d_in[4];
    const float* b2m   = (const float*)d_in[5];
    const float* W1u   = (const float*)d_in[6];
    const float* b1u   = (const float*)d_in[7];
    const float* W2u   = (const float*)d_in[8];
    const float* b2u   = (const float*)d_in[9];
    const float* gamma = (const float*)d_in[10];
    const float* beta  = (const float*)d_in[11];

    _Float16* ws     = (_Float16*)d_ws;
    _Float16* W1m_f  = ws;               // 256x128 halves
    _Float16* W2m_f  = ws + 32768;       // 128x128
    _Float16* W1u_f  = ws + 49152;       // 256x128
    _Float16* W2u_f  = ws + 81920;       // 128x128  (total 196608 B)

    pack_w<<<128, 256, 0, stream>>>(W1m, W1m_f, 32768);
    pack_w<<< 64, 256, 0, stream>>>(W2m, W2m_f, 16384);
    pack_w<<<128, 256, 0, stream>>>(W1u, W1u_f, 32768);
    pack_w<<< 64, 256, 0, stream>>>(W2u, W2u_f, 16384);

    const size_t shmem = 84480;          // 84.5 KB dynamic LDS (WGP has 320 KB)
    gnn_layer<<<B_ * NTILES_PB, 256, shmem, stream>>>(
        h, tour, W1m_f, b1m, W2m_f, b2m, W1u_f, b1u, W2u_f, b2u,
        gamma, beta, (float*)d_out);
}
